// AlignBlock_12197707121005
// MI455X (gfx1250) — matrix operand order repaired and checked by reading, not compiled, not run on hardware
//
#include <hip/hip_runtime.h>
#include <math.h>

#define HH 128
#define WW 128
#define BB 4

typedef __attribute__((ext_vector_type(16))) __bf16          v16bf;
typedef __attribute__((ext_vector_type(8)))  float           v8f;
typedef __attribute__((ext_vector_type(8)))  unsigned short  su8;

union BFu { v16bf v; su8 h[2]; unsigned short u[16]; };

__device__ __forceinline__ unsigned short f2bf(float f) {
  union { float f; unsigned u; } x; x.f = f;
  unsigned u = x.u;
  u += 0x7FFFu + ((u >> 16) & 1u);       // round-to-nearest-even
  return (unsigned short)(u >> 16);
}
__device__ __forceinline__ float bf2f(unsigned short h) {
  union { float f; unsigned u; } x; x.u = ((unsigned)h) << 16;
  return x.f;
}

// f32 -> bf16 elementwise, 4 elems/thread, n must be a multiple of 4*blockDim
__global__ void cvt_bf16(const float* __restrict__ in, unsigned short* __restrict__ out) {
  const int i0 = (blockIdx.x * blockDim.x + threadIdx.x) * 4;
#pragma unroll
  for (int i = 0; i < 4; ++i) out[i0 + i] = f2bf(in[i0 + i]);
}

// OIHW f32 weights -> bf16 tap-major [Cout][t*Cin + c] (the 3x3 GEMM K order)
__global__ void prep_w(const float* __restrict__ in, unsigned short* __restrict__ out,
                       int Cout, int Cin) {
  const int idx = blockIdx.x * blockDim.x + threadIdx.x;
  const int Kt = Cin * 9;
  if (idx >= Cout * Kt) return;
  const int m = idx / Kt, r = idx % Kt;
  const int t = r / Cin, c = r % Cin;
  out[idx] = f2bf(in[((size_t)m * Cin + c) * 9 + t]);
}

// Implicit-GEMM conv3x3 (pad=1) over bf16 activations, or 1x1 GEMM over the
// K-fastest im2col buffer (onexone=1: in0 layout [b][hw][Cin] bf16).
//   wT    : bf16 weights, [Cout][9*Cin] tap-major (3x3) or [Cout][Cin] (1x1)
//   mode  : 1 = LeakyReLU(0.1), 2 = sigmoid for channels >= 144
// Tile M=64 x N=64 x K=32, 8 wave32s; 2x v_wmma_f32_16x16x32_bf16 per wave/K-step.
// B tile transposed in LDS ([N][K], 16B rows) so fragment loads are ds_load_b128.
// 1x1 path: double-buffered global_load_async_to_lds_b128 pipeline, waiting
// s_wait_asynccnt 0x1 so the next chunk's DMA overlaps the WMMAs.
__global__ void conv_wmma(const unsigned short* __restrict__ in0,
                          const unsigned short* __restrict__ in1,
                          const unsigned short* __restrict__ wT,
                          const float* __restrict__ bias,
                          float* __restrict__ outF,
                          unsigned short* __restrict__ outH,
                          int Cin, int Cout, int mode, int onexone)
{
  __shared__ __align__(16) unsigned short lA [64][40];     // [M][K] weights
  __shared__ __align__(16) unsigned short lBt[2][64][40];  // [N][K] acts, 2 bufs

  const int tid    = threadIdx.x;
  const int lane   = tid & 31;            // wave32
  const int wid    = tid >> 5;
  const int wave_m = wid & 3;             // 4 M sub-tiles of 16
  const int wave_n = wid >> 2;            // 2 N sub-tiles of 32

  const int n0  = blockIdx.x * 64;        // 64-pixel tile (one row segment)
  const int b   = n0 >> 14;
  const int hw0 = n0 & 16383;
  const int y   = hw0 >> 7;
  const int xp  = hw0 & 127;
  const int m0  = blockIdx.y * 64;        // Cout tile

  const int cchunks = Cin >> 5;
  const int nchunks = onexone ? cchunks : 9 * cchunks;
  const int Ktot    = onexone ? Cin : Cin * 9;

  auto issue_async = [&](int qq, int bufsel) {
    const int n   = tid >> 2;             // 0..63
    const int seg = tid & 3;              // 4 x 8 u16 segments
    const unsigned short* gp =
        in0 + ((size_t)b * 16384 + hw0 + n) * 576 + (qq << 5) + seg * 8;
    const unsigned la = (unsigned)(size_t)&lBt[bufsel][n][seg * 8];
    asm volatile("global_load_async_to_lds_b128 %0, %1, off"
                 :: "v"(la), "v"((unsigned long long)(size_t)gp)
                 : "memory");
  };

  if (onexone) issue_async(0, 0);

  v8f acc0 = {}; v8f acc1 = {};

  for (int q = 0; q < nchunks; ++q) {
    const int t     = onexone ? 0 : (q / cchunks);          // tap (tap-major K)
    const int cbase = (onexone ? q : (q % cchunks)) << 5;
    const int kbase = onexone ? cbase : t * Cin + cbase;

    // ---- A tile: 64(M) x 32(K) bf16 weights, one b128 load + store ----
    {
      const int m  = tid >> 2;
      const int kp = (tid & 3) << 3;
      const int mg = m0 + m;
      su8 av = {};
      if (mg < Cout) av = *(const su8*)&wT[(size_t)mg * Ktot + kbase + kp];
      *(su8*)&lA[m][kp] = av;
    }
    // ---- B tile (transposed [N][K]) ----
    if (onexone) {
      if (q + 1 < nchunks) {
        issue_async(q + 1, (q + 1) & 1);                    // overlap next DMA
        asm volatile("s_wait_asynccnt 0x1" ::: "memory");   // chunk q resident
      } else {
        asm volatile("s_wait_asynccnt 0x0" ::: "memory");
      }
    } else {
      const int dy = t / 3 - 1;
      const int dx = t % 3 - 1;
      const int kl = tid >> 3;            // channel within chunk
      const int np = (tid & 7) << 3;      // 8 pixels per thread
      const int c  = cbase + kl;
      const int yy = y + dy;
      const unsigned short* src = in0; int cc = c; int Ct = Cin;
      if (in1) { Ct = 64; if (c >= 64) { src = in1; cc = c - 64; } }
      const bool yok = (yy >= 0) && (yy < HH);
      const size_t rowbase = (((size_t)b * Ct + cc) * HH + (yok ? yy : 0)) * WW;
#pragma unroll
      for (int i = 0; i < 8; ++i) {
        const int xx = xp + np + i + dx;
        unsigned short v = 0;             // bf16 +0.0
        if (yok && xx >= 0 && xx < WW) v = src[rowbase + xx];
        lBt[0][np + i][kl] = v;
      }
    }
    __syncthreads();

    // ---- fragments: all 16B LDS vector loads ----
    const unsigned short (*Bt)[40] = lBt[onexone ? (q & 1) : 0];
    BFu af, bf0, bf1;
    const int mrow = wave_m * 16 + (lane & 15);
    const int hi   = (lane >> 4) & 1;
    af.h[0]  = *(const su8*)&lA[mrow][hi * 8];        // K 0-7 | 8-15
    af.h[1]  = *(const su8*)&lA[mrow][hi * 8 + 16];   // K 16-23 | 24-31
    const int nr0 = wave_n * 32 + (lane & 15);
    const int nr1 = nr0 + 16;
    bf0.h[0] = *(const su8*)&Bt[nr0][hi * 8];         // K 0-7 | 8-15
    bf0.h[1] = *(const su8*)&Bt[nr0][hi * 8 + 16];    // K 16-23 | 24-31
    bf1.h[0] = *(const su8*)&Bt[nr1][hi * 8];
    bf1.h[1] = *(const su8*)&Bt[nr1][hi * 8 + 16];

    acc0 = __builtin_amdgcn_wmma_f32_16x16x32_bf16(false, af.v, false, bf0.v,
                                                   (short)0, acc0, false, false);
    acc1 = __builtin_amdgcn_wmma_f32_16x16x32_bf16(false, af.v, false, bf1.v,
                                                   (short)0, acc1, false, false);
    __syncthreads();
  }

  // ---- epilogue: bias + activation, store f32 and/or bf16 ----
  const int nloc  = lane & 15;
  const int mside = (lane < 16) ? 0 : 8;
#pragma unroll
  for (int r = 0; r < 8; ++r) {
    const int m = m0 + wave_m * 16 + mside + r;
    if (m >= Cout) continue;
    const float bv = bias[m];
    float v0 = acc0[r] + bv;
    float v1 = acc1[r] + bv;
    if (mode == 1) {
      v0 = v0 >= 0.f ? v0 : 0.1f * v0;
      v1 = v1 >= 0.f ? v1 : 0.1f * v1;
    } else if (mode == 2 && m >= 144) {
      v0 = 1.f / (1.f + __expf(-v0));
      v1 = 1.f / (1.f + __expf(-v1));
    }
    const size_t o = (((size_t)b * Cout + m) * HH + y) * WW + xp + wave_n * 32;
    if (outF) { outF[o + nloc] = v0;       outF[o + 16 + nloc] = v1; }
    if (outH) { outH[o + nloc] = f2bf(v0); outH[o + 16 + nloc] = f2bf(v1); }
  }
}

// Modulated deformable im2col over bf16 features. One thread per
// (b, group, pixel): 9 taps x 8 channels -> contiguous 144B slice of
// cols[b][hw][K=576] (K = c*9 + tap), written as 9 aligned 16B stores.
__global__ void deform_im2col(const unsigned short* __restrict__ x16,
                              const float* __restrict__ om,
                              unsigned short* __restrict__ cols)
{
  const int idx = blockIdx.x * blockDim.x + threadIdx.x;
  const int hw = idx & 16383;
  const int g  = (idx >> 14) & 7;
  const int b  = idx >> 17;
  if (b >= BB) return;
  const int y  = hw >> 7;
  const int xq = hw & 127;

  const size_t obase = (size_t)b * 216 * 16384 + hw;
  const unsigned short* xb = x16 + (size_t)b * 64 * 16384;

  unsigned short vals[72];
#pragma unroll
  for (int k = 0; k < 9; ++k) {
    const float oy = om[obase + (size_t)(g * 18 + 2 * k)     * 16384];
    const float ox = om[obase + (size_t)(g * 18 + 2 * k + 1) * 16384];
    const float mk = om[obase + (size_t)(144 + g * 9 + k)    * 16384]; // sigmoided

    const float py = (float)y  + (float)(k / 3 - 1) + oy;
    const float px = (float)xq + (float)(k % 3 - 1) + ox;
    const float fy = floorf(py), fx = floorf(px);
    const int y0 = (int)fy, x0 = (int)fx;
    const float ly = py - fy, lx = px - fx;

    const int y0c = min(max(y0, 0), HH - 1);
    const int y1c = min(max(y0 + 1, 0), HH - 1);
    const int x0c = min(max(x0, 0), WW - 1);
    const int x1c = min(max(x0 + 1, 0), WW - 1);
    const bool yv0 = (y0 >= 0) && (y0 < HH);
    const bool yv1 = (y0 + 1 >= 0) && (y0 + 1 < HH);
    const bool xv0 = (x0 >= 0) && (x0 < WW);
    const bool xv1 = (x0 + 1 >= 0) && (x0 + 1 < WW);
    const float w00 = (yv0 && xv0) ? (1.f - ly) * (1.f - lx) : 0.f;
    const float w01 = (yv0 && xv1) ? (1.f - ly) * lx         : 0.f;
    const float w10 = (yv1 && xv0) ? ly * (1.f - lx)         : 0.f;
    const float w11 = (yv1 && xv1) ? ly * lx                 : 0.f;

    const int i00 = y0c * WW + x0c, i01 = y0c * WW + x1c;
    const int i10 = y1c * WW + x0c, i11 = y1c * WW + x1c;

#pragma unroll
    for (int c = 0; c < 8; ++c) {
      const unsigned short* xc = xb + (size_t)(g * 8 + c) * 16384;
      const float v = w00 * bf2f(xc[i00]) + w01 * bf2f(xc[i01])
                    + w10 * bf2f(xc[i10]) + w11 * bf2f(xc[i11]);
      vals[c * 9 + k] = f2bf(v * mk);
    }
  }
  unsigned short* dst = cols + ((size_t)b * 16384 + hw) * 576 + g * 72;
#pragma unroll
  for (int s = 0; s < 9; ++s)
    *(su8*)&dst[s * 8] = *(const su8*)&vals[s * 8];
}

extern "C" void kernel_launch(void* const* d_in, const int* in_sizes, int n_in,
                              void* d_out, int out_size, void* d_ws, size_t ws_size,
                              hipStream_t stream) {
  const float* fea   = (const float*)d_in[0];
  const float* ref   = (const float*)d_in[1];
  const float* w_of1 = (const float*)d_in[2];  const float* b_of1 = (const float*)d_in[3];
  const float* w_of2 = (const float*)d_in[4];  const float* b_of2 = (const float*)d_in[5];
  const float* w_or1 = (const float*)d_in[6];  const float* b_or1 = (const float*)d_in[7];
  const float* w_or2 = (const float*)d_in[8];  const float* b_or2 = (const float*)d_in[9];
  const float* w_om  = (const float*)d_in[10]; const float* b_om  = (const float*)d_in[11];
  const float* w_dcn = (const float*)d_in[12]; const float* b_dcn = (const float*)d_in[13];
  const float* w_f1  = (const float*)d_in[14]; const float* b_f1  = (const float*)d_in[15];
  const float* w_f2  = (const float*)d_in[16]; const float* b_f2  = (const float*)d_in[17];

  char* ws = (char*)d_ws;
  typedef unsigned short u16;
  const size_t S16   = (size_t)4 * 64 * 16384 * 2;   // 8 MB bf16 [4,64,128,128]
  const size_t SOM   = (size_t)4 * 216 * 16384 * 4;  // 54 MB f32 [4,216,128,128]
  const size_t SCOLS = (size_t)4 * 16384 * 576 * 2;  // 72 MB bf16 [4,16384,576]
  u16*   FEA16 = (u16*)ws;
  u16*   REF16 = (u16*)(ws + 1 * S16);
  u16*   OF1h  = (u16*)(ws + 2 * S16);
  u16*   OF2h  = (u16*)(ws + 3 * S16);
  u16*   DFh   = (u16*)(ws + 4 * S16);
  u16*   DRh   = (u16*)(ws + 5 * S16);
  float* OM    = (float*)(ws + 6 * S16);
  u16*   COLS  = (u16*)(ws + 6 * S16 + SOM);
  char*  wsw   = ws + 6 * S16 + SOM + SCOLS;         // bf16 weight area (~1 MB)
  u16* WT_OF1 = (u16*)wsw;                wsw += 73728 * 2;
  u16* WT_OF2 = (u16*)wsw;                wsw += 36864 * 2;
  u16* WT_OR1 = (u16*)wsw;                wsw += 73728 * 2;
  u16* WT_OR2 = (u16*)wsw;                wsw += 36864 * 2;
  u16* WT_OM  = (u16*)wsw;                wsw += 124416 * 2;
  u16* WT_DCN = (u16*)wsw;                wsw += 36864 * 2;
  u16* WT_F1  = (u16*)wsw;                wsw += 73728 * 2;
  u16* WT_F2  = (u16*)wsw;
  u16*   F1h   = OF1h;                               // safe to reuse at fusion time

  const dim3 blk(256);
  const dim3 g64(1024, 1);   // 65536 px / 64-px tiles
  const dim3 gOM(1024, 4);   // Cout=216 -> 4 guarded M-tiles
  const dim3 gdef(2048);     // 4*8*16384 / 256

  // --- one-time precision/layout prep (bf16 activations + tap-major weights) ---
  cvt_bf16<<<dim3(4096), blk, 0, stream>>>(fea, FEA16);
  cvt_bf16<<<dim3(4096), blk, 0, stream>>>(ref, REF16);
  cvt_bf16<<<dim3(36),   blk, 0, stream>>>(w_dcn, WT_DCN);   // [64][576] natural K order
  prep_w<<<dim3(288), blk, 0, stream>>>(w_of1, WT_OF1, 64, 128);
  prep_w<<<dim3(144), blk, 0, stream>>>(w_of2, WT_OF2, 64, 64);
  prep_w<<<dim3(288), blk, 0, stream>>>(w_or1, WT_OR1, 64, 128);
  prep_w<<<dim3(144), blk, 0, stream>>>(w_or2, WT_OR2, 64, 64);
  prep_w<<<dim3(486), blk, 0, stream>>>(w_om,  WT_OM, 216, 64);
  prep_w<<<dim3(288), blk, 0, stream>>>(w_f1,  WT_F1, 64, 128);
  prep_w<<<dim3(144), blk, 0, stream>>>(w_f2,  WT_F2, 64, 64);

  // --- fea branch ---
  conv_wmma<<<g64, blk, 0, stream>>>(FEA16, REF16, WT_OF1, b_of1, nullptr, OF1h, 128, 64, 1, 0);
  conv_wmma<<<g64, blk, 0, stream>>>(OF1h, nullptr, WT_OF2, b_of2, nullptr, OF2h, 64, 64, 1, 0);
  conv_wmma<<<gOM, blk, 0, stream>>>(OF2h, nullptr, WT_OM, b_om, OM, nullptr, 64, 216, 2, 0);
  deform_im2col<<<gdef, blk, 0, stream>>>(FEA16, OM, COLS);
  conv_wmma<<<g64, blk, 0, stream>>>(COLS, nullptr, WT_DCN, b_dcn, nullptr, DFh, 576, 64, 1, 1);
  // --- ref branch (shared DCN weights) ---
  conv_wmma<<<g64, blk, 0, stream>>>(REF16, FEA16, WT_OR1, b_or1, nullptr, OF1h, 128, 64, 1, 0);
  conv_wmma<<<g64, blk, 0, stream>>>(OF1h, nullptr, WT_OR2, b_or2, nullptr, OF2h, 64, 64, 1, 0);
  conv_wmma<<<gOM, blk, 0, stream>>>(OF2h, nullptr, WT_OM, b_om, OM, nullptr, 64, 216, 2, 0);
  deform_im2col<<<gdef, blk, 0, stream>>>(REF16, OM, COLS);
  conv_wmma<<<g64, blk, 0, stream>>>(COLS, nullptr, WT_DCN, b_dcn, nullptr, DRh, 576, 64, 1, 1);
  // --- fusion ---
  conv_wmma<<<g64, blk, 0, stream>>>(DFh, DRh, WT_F1, b_f1, nullptr, F1h, 128, 64, 1, 0);
  conv_wmma<<<g64, blk, 0, stream>>>(F1h, nullptr, WT_F2, b_f2, (float*)d_out, nullptr, 64, 64, 1, 0);
}
